// DCN_75539884802539
// MI455X (gfx1250) — matrix operand order repaired and checked by reading, not compiled, not run on hardware
//
#include <hip/hip_runtime.h>
#include <hip/hip_bf16.h>
#include <math.h>

typedef __attribute__((ext_vector_type(16))) __bf16 v16bf;
typedef __attribute__((ext_vector_type(8)))  float  v8f;

#define B_ 4
#define C_ 64
#define H_ 128
#define W_ 128

__device__ __forceinline__ unsigned short f2bfbits(float f) {
    union { float f; unsigned u; } a; a.f = f;
    unsigned r = a.u + 0x7FFFu + ((a.u >> 16) & 1u);
    return (unsigned short)(r >> 16);
}
__device__ __forceinline__ __bf16 bits2bf(unsigned short s) {
    union { unsigned short s; __bf16 b; } c; c.s = s; return c.b;
}
__device__ __forceinline__ int iclamp(int v, int lo, int hi) {
    return v < lo ? lo : (v > hi ? hi : v);
}

// ---------------------------------------------------------------------------
// Pack an OIHW f32 weight [M][Cin*9] into WMMA A-fragment order (bf16):
// packed[((mt*KC + kc)*32 + lane)*16 + j], contiguous k order (B gathered the same way):
//   lane<16  -> M=mt*16+lane,    K=kc*32+j
//   lane>=16 -> M=mt*16+lane-16, K=kc*32+16+j
// ---------------------------------------------------------------------------
__global__ void pack_w(const float* __restrict__ w, unsigned short* __restrict__ pw,
                       int Mvalid, int Ktot, int KC, int total) {
    int i = blockIdx.x * 256 + threadIdx.x;
    if (i >= total) return;
    int j    = i & 15;
    int lane = (i >> 4) & 31;
    int ck   = i >> 9;
    int kc   = ck % KC;
    int mt   = ck / KC;
    int m    = mt * 16 + (lane & 15);
    int kpos = ((lane < 16) ? 0 : 16) + j;
    int K    = kc * 32 + kpos;
    float v  = (m < Mvalid) ? w[m * Ktot + K] : 0.f;
    pw[i] = f2bfbits(v);
}

// ---------------------------------------------------------------------------
// Bicubic x2 upsample (align-corners style: src = o*(n-1)/(no-1)), A=-0.75
// ---------------------------------------------------------------------------
__device__ __forceinline__ float cubicw(float t) {
    const float A = -0.75f;
    float t2 = t * t, t3 = t2 * t;
    float w1 = (A + 2.f) * t3 - (A + 3.f) * t2 + 1.f;
    float w2 = A * t3 - 5.f * A * t2 + 8.f * A * t - 4.f * A;
    return (t <= 1.f) ? w1 : ((t < 2.f) ? w2 : 0.f);
}

__global__ void up2_bicubic(const float* __restrict__ in, float* __restrict__ out,
                            int total, float mult) {
    int i = blockIdx.x * 256 + threadIdx.x;
    if (i >= total) return;
    int ox = i & 127;
    int oy = (i >> 7) & 127;
    int bc = i >> 14;
    const float s = 63.f / 127.f;
    float sy = oy * s, sx = ox * s;
    int y0 = (int)floorf(sy), x0 = (int)floorf(sx);
    const float* p = in + bc * 64 * 64;
    float acc = 0.f;
    #pragma unroll
    for (int ty = 0; ty < 4; ++ty) {
        int yi = y0 - 1 + ty;
        float wy = cubicw(fabsf(sy - (float)yi));
        int yc = iclamp(yi, 0, 63);
        #pragma unroll
        for (int tx = 0; tx < 4; ++tx) {
            int xi = x0 - 1 + tx;
            float wx = cubicw(fabsf(sx - (float)xi));
            int xc = iclamp(xi, 0, 63);
            acc += wy * wx * p[yc * 64 + xc];
        }
    }
    out[i] = acc * mult;
}

// f32 ref features -> bf16 into channels [64..127] of the concat buffer
__global__ void cvt_ref(const float* __restrict__ ref, unsigned short* __restrict__ xcat) {
    int i = blockIdx.x * 256 + threadIdx.x;
    if (i >= B_ * C_ * H_ * W_) return;
    int px = i & (H_ * W_ - 1);
    int c  = (i >> 14) & 63;
    int b  = i >> 20;
    xcat[((b * 128 + 64 + c) << 14) + px] = f2bfbits(ref[i]);
}

// ---------------------------------------------------------------------------
// Fused modulated deformable conv (gather + WMMA GEMM), C=64, K taps=9.
// One workgroup per (b,h) row: 8 waves x 16 pixels, 4 M-tiles of 16 channels.
// FINAL=0: write bf16 (no act) into 128-ch concat buffer channels 0..63
// FINAL=1: write f32 lrelu into d_out
// ---------------------------------------------------------------------------
template <int FINAL>
__global__ void dcn_wmma(const float* __restrict__ x,            // [B,64,H,W]
                         const float* __restrict__ off,          // [B,18,H,W]
                         const float* __restrict__ msk,          // [B,9,H,W]
                         const unsigned short* __restrict__ pw,  // packed A [4][18][512]
                         const float* __restrict__ bias,
                         unsigned short* __restrict__ out_bf,
                         float* __restrict__ out_f) {
    __shared__ float4   s_w[9 * 128];
    __shared__ unsigned s_c[9 * 128];

    int wg = blockIdx.x;
    int b = wg >> 7, h = wg & 127;
    int tid = threadIdx.x;
    int lane = tid & 31, wv = tid >> 5;

    // Phase 1: per-(tap,pixel) bilinear corner coords + mask-folded weights
    for (int e = tid; e < 9 * 128; e += 256) {
        int n = e & 127, kk = e >> 7;
        int ky = kk / 3, kx = kk % 3;
        float dy = off[((b * 18 + 2 * kk + 0) << 14) + (h << 7) + n];
        float dx = off[((b * 18 + 2 * kk + 1) << 14) + (h << 7) + n];
        float m  = msk[((b * 9 + kk) << 14) + (h << 7) + n];
        float py = (float)(h - 1 + ky) + dy;
        float px = (float)(n - 1 + kx) + dx;
        float y0f = floorf(py), x0f = floorf(px);
        float ty = py - y0f, tx = px - x0f;
        int y0 = (int)y0f, x0 = (int)x0f;
        int y1 = y0 + 1,  x1 = x0 + 1;
        float vy0 = (y0 >= 0 && y0 < H_) ? 1.f : 0.f;
        float vy1 = (y1 >= 0 && y1 < H_) ? 1.f : 0.f;
        float vx0 = (x0 >= 0 && x0 < W_) ? 1.f : 0.f;
        float vx1 = (x1 >= 0 && x1 < W_) ? 1.f : 0.f;
        float4 w4;
        w4.x = (1.f - ty) * (1.f - tx) * vy0 * vx0 * m;
        w4.y = (1.f - ty) * tx         * vy0 * vx1 * m;
        w4.z = ty * (1.f - tx)         * vy1 * vx0 * m;
        w4.w = ty * tx                 * vy1 * vx1 * m;
        int cy0 = iclamp(y0, 0, H_ - 1), cy1 = iclamp(y1, 0, H_ - 1);
        int cx0 = iclamp(x0, 0, W_ - 1), cx1 = iclamp(x1, 0, W_ - 1);
        s_w[e] = w4;
        s_c[e] = (unsigned)cy0 | ((unsigned)cx0 << 8) | ((unsigned)cy1 << 16) | ((unsigned)cx1 << 24);
    }
    __syncthreads();

    v8f acc0 = {}, acc1 = {}, acc2 = {}, acc3 = {};
    int n = (wv << 4) + (lane & 15);
    int kbase = (lane < 16) ? 0 : 16;
    const float* xb = x + (b * 64) * (H_ * W_);

    for (int kc = 0; kc < 18; ++kc) {
        v16bf bfrag;
        #pragma unroll
        for (int j = 0; j < 16; ++j) {
            int K = kc * 32 + kbase + j;
            int c = K / 9, kk = K % 9;
            int e = (kk << 7) + n;
            float4 w4 = s_w[e];
            unsigned cc = s_c[e];
            int cy0 = cc & 255, cx0 = (cc >> 8) & 255;
            int cy1 = (cc >> 16) & 255, cx1 = (cc >> 24) & 255;
            const float* xc = xb + (c << 14);
            float val = w4.x * xc[(cy0 << 7) + cx0]
                      + w4.y * xc[(cy0 << 7) + cx1]
                      + w4.z * xc[(cy1 << 7) + cx0]
                      + w4.w * xc[(cy1 << 7) + cx1];
            bfrag[j] = bits2bf(f2bfbits(val));
        }
        const unsigned short* pk = pw + (kc << 9) + (lane << 4);
        v16bf a0 = *(const v16bf*)(pk + 0 * 18 * 512);
        v16bf a1 = *(const v16bf*)(pk + 1 * 18 * 512);
        v16bf a2 = *(const v16bf*)(pk + 2 * 18 * 512);
        v16bf a3 = *(const v16bf*)(pk + 3 * 18 * 512);
        acc0 = __builtin_amdgcn_wmma_f32_16x16x32_bf16(false, a0, false, bfrag, (short)0, acc0, false, false);
        acc1 = __builtin_amdgcn_wmma_f32_16x16x32_bf16(false, a1, false, bfrag, (short)0, acc1, false, false);
        acc2 = __builtin_amdgcn_wmma_f32_16x16x32_bf16(false, a2, false, bfrag, (short)0, acc2, false, false);
        acc3 = __builtin_amdgcn_wmma_f32_16x16x32_bf16(false, a3, false, bfrag, (short)0, acc3, false, false);
    }

    int mbase = (lane < 16) ? 0 : 8;
    #pragma unroll
    for (int mt = 0; mt < 4; ++mt) {
        v8f acc = (mt == 0) ? acc0 : (mt == 1) ? acc1 : (mt == 2) ? acc2 : acc3;
        #pragma unroll
        for (int r = 0; r < 8; ++r) {
            int M = mt * 16 + mbase + r;
            float v = acc[r] + bias[M];
            if (FINAL) {
                v = (v >= 0.f) ? v : 0.1f * v;
                out_f[((b * 64 + M) << 14) + (h << 7) + n] = v;
            } else {
                out_bf[((b * 128 + M) << 14) + (h << 7) + n] = f2bfbits(v);
            }
        }
    }
}

// ---------------------------------------------------------------------------
// Implicit-GEMM conv3x3 (pad 1) with WMMA. One workgroup per (b,h) output row.
// Input is bf16 [B][Ctot][H][W]; K loop runs over cin_blocks blocks of 64
// channels, staging 64ch x 3 rows x 130 cols in LDS per block.
// MODE 0: bias + lrelu -> bf16 out (64 channels)
// MODE 2: offset head (Mvalid=27): ch<18 -> 8*tanh+flow_up -> out_off,
//         ch 18..26 -> sigmoid -> out_m
// ---------------------------------------------------------------------------
template <int MTILES, int MODE>
__global__ void conv3x3_wmma(const unsigned short* __restrict__ in,
                             const unsigned short* __restrict__ pw,
                             const float* __restrict__ bias,
                             int cin_blocks, int Ctot,
                             unsigned short* __restrict__ out_bf,
                             float* __restrict__ out_off,
                             float* __restrict__ out_m,
                             const float* __restrict__ flow_up) {
    __shared__ unsigned short tile[64 * 3 * 130];  // ~50 KB

    int wg = blockIdx.x;
    int b = wg >> 7, h = wg & 127;
    int tid = threadIdx.x;
    int lane = tid & 31, wv = tid >> 5;
    int n = (wv << 4) + (lane & 15);
    int kbase = (lane < 16) ? 0 : 16;
    int KCtot = cin_blocks * 18;

    v8f acc[MTILES];
    #pragma unroll
    for (int mt = 0; mt < MTILES; ++mt) acc[mt] = (v8f){};

    for (int cb = 0; cb < cin_blocks; ++cb) {
        __syncthreads();
        for (int e = tid; e < 64 * 3 * 130; e += 256) {
            int col = e % 130;
            int t   = e / 130;
            int ry  = t % 3;
            int c   = t / 3;
            int gx = col - 1;
            int gy = h - 1 + ry;
            unsigned short v = 0;
            if (gx >= 0 && gx < W_ && gy >= 0 && gy < H_)
                v = in[((b * Ctot + cb * 64 + c) << 14) + (gy << 7) + gx];
            tile[c * 390 + ry * 130 + col] = v;
        }
        __syncthreads();

        for (int lc = 0; lc < 18; ++lc) {
            int kc = cb * 18 + lc;
            v16bf bfrag;
            #pragma unroll
            for (int j = 0; j < 16; ++j) {
                int K = lc * 32 + kbase + j;
                int c = K / 9, r = K % 9;
                int ky = r / 3, kx = r % 3;
                bfrag[j] = bits2bf(tile[c * 390 + ky * 130 + (n + kx)]);
            }
            #pragma unroll
            for (int mt = 0; mt < MTILES; ++mt) {
                v16bf a = *(const v16bf*)(pw + (((mt * KCtot + kc) << 5) + lane) * 16);
                acc[mt] = __builtin_amdgcn_wmma_f32_16x16x32_bf16(false, a, false, bfrag,
                                                                  (short)0, acc[mt], false, false);
            }
        }
    }

    int mbase = (lane < 16) ? 0 : 8;
    #pragma unroll
    for (int mt = 0; mt < MTILES; ++mt) {
        #pragma unroll
        for (int r = 0; r < 8; ++r) {
            int M = mt * 16 + mbase + r;
            float v = acc[mt][r];
            if (MODE == 0) {
                v += bias[M];
                v = (v >= 0.f) ? v : 0.1f * v;
                out_bf[((b * 64 + M) << 14) + (h << 7) + n] = f2bfbits(v);
            } else {
                if (M < 18) {
                    v += bias[M];
                    float f = 8.f * tanhf(v) + flow_up[((b * 18 + M) << 14) + (h << 7) + n];
                    out_off[((b * 18 + M) << 14) + (h << 7) + n] = f;
                } else if (M < 27) {
                    v += bias[M];
                    out_m[((b * 9 + (M - 18)) << 14) + (h << 7) + n] = 1.f / (1.f + expf(-v));
                }
            }
        }
    }
}

// ---------------------------------------------------------------------------

extern "C" void kernel_launch(void* const* d_in, const int* in_sizes, int n_in,
                              void* d_out, int out_size, void* d_ws, size_t ws_size,
                              hipStream_t stream) {
    const float* nbr    = (const float*)d_in[0];
    const float* ref    = (const float*)d_in[1];
    const float* flow   = (const float*)d_in[2];
    const float* mask   = (const float*)d_in[3];
    const float* w1     = (const float*)d_in[4];
    const float* b1     = (const float*)d_in[5];
    const float* w2     = (const float*)d_in[6];
    const float* b2     = (const float*)d_in[7];
    const float* w3     = (const float*)d_in[8];
    const float* b3     = (const float*)d_in[9];
    const float* w_off  = (const float*)d_in[10];
    const float* b_off  = (const float*)d_in[11];
    const float* w_dcn  = (const float*)d_in[12];
    const float* b_dcn  = (const float*)d_in[13];
    const float* w_zero = (const float*)d_in[14];
    const float* b_zero = (const float*)d_in[15];

    char* ws = (char*)d_ws;
    float*          flow_up = (float*)(ws + 0);                 //  4,718,592 B
    float*          mask_up = (float*)(ws + 4718592);           //  2,359,296 B
    unsigned short* xcat    = (unsigned short*)(ws + 7077888);  // 16,777,216 B (bf16, 128ch)
    unsigned short* feat1   = (unsigned short*)(ws + 23855104); //  8,388,608 B
    unsigned short* feat2   = (unsigned short*)(ws + 32243712);
    unsigned short* feat3   = (unsigned short*)(ws + 40632320);
    unsigned short* pw1     = (unsigned short*)(ws + 49020928); // 147,456 B
    unsigned short* pw2     = (unsigned short*)(ws + 49168384); //  73,728 B
    unsigned short* pw3     = (unsigned short*)(ws + 49242112);
    unsigned short* pwoff   = (unsigned short*)(ws + 49315840); //  36,864 B
    unsigned short* pwdcn   = (unsigned short*)(ws + 49352704);
    unsigned short* pwz     = (unsigned short*)(ws + 49426432);

    float* out_y   = (float*)d_out;                 // [B,64,H,W] lrelu(y)
    float* out_off = out_y + 4 * 64 * 128 * 128;    // [B,18,H,W] offset
    float* out_m   = out_off + 4 * 18 * 128 * 128;  // [B,9,H,W]  mask

    // Pack all weights into WMMA A-fragment order (bf16)
    pack_w<<<(73728 + 255) / 256, 256, 0, stream>>>(w1,     pw1,   64, 1152, 36, 73728);
    pack_w<<<(36864 + 255) / 256, 256, 0, stream>>>(w2,     pw2,   64,  576, 18, 36864);
    pack_w<<<(36864 + 255) / 256, 256, 0, stream>>>(w3,     pw3,   64,  576, 18, 36864);
    pack_w<<<(18432 + 255) / 256, 256, 0, stream>>>(w_off,  pwoff, 27,  576, 18, 18432);
    pack_w<<<(36864 + 255) / 256, 256, 0, stream>>>(w_dcn,  pwdcn, 64,  576, 18, 36864);
    pack_w<<<(36864 + 255) / 256, 256, 0, stream>>>(w_zero, pwz,   64,  576, 18, 36864);

    // Bicubic x2 upsample: flow (*2.0) and mask (*1.0)
    up2_bicubic<<<(1179648 + 255) / 256, 256, 0, stream>>>(flow, flow_up, 1179648, 2.0f);
    up2_bicubic<<<(589824 + 255) / 256, 256, 0, stream>>>(mask, mask_up, 589824, 1.0f);

    // ref features -> bf16 into concat buffer channels 64..127
    cvt_ref<<<(4194304 + 255) / 256, 256, 0, stream>>>(ref, xcat);

    // warp = DCN(nbr, flow_up, mask_up, w_zero) -> concat channels 0..63 (bf16)
    dcn_wmma<0><<<512, 256, 0, stream>>>(nbr, flow_up, mask_up, pwz, b_zero, xcat, nullptr);

    // feature trunk
    conv3x3_wmma<4, 0><<<512, 256, 0, stream>>>(xcat,  pw1, b1, 2, 128, feat1, nullptr, nullptr, nullptr);
    conv3x3_wmma<4, 0><<<512, 256, 0, stream>>>(feat1, pw2, b2, 1,  64, feat2, nullptr, nullptr, nullptr);
    conv3x3_wmma<4, 0><<<512, 256, 0, stream>>>(feat2, pw3, b3, 1,  64, feat3, nullptr, nullptr, nullptr);

    // offset head: 8*tanh + flow_up -> out_off, sigmoid -> out_m
    conv3x3_wmma<2, 2><<<512, 256, 0, stream>>>(feat3, pwoff, b_off, 1, 64, nullptr, out_off, out_m, flow_up);

    // final DCN with learned offsets/mask, lrelu -> out_y
    dcn_wmma<1><<<512, 256, 0, stream>>>(nbr, out_off, out_m, pwdcn, b_dcn, nullptr, out_y);
}
